// BaselineLSTM_44607530336396
// MI455X (gfx1250) — matrix operand order repaired and checked by reading, not compiled, not run on hardware
//
#include <hip/hip_runtime.h>

typedef _Float16 v16h __attribute__((ext_vector_type(16)));
typedef _Float16 v8h  __attribute__((ext_vector_type(8)));
typedef float    v8f  __attribute__((ext_vector_type(8)));
typedef unsigned int u32x4 __attribute__((ext_vector_type(4)));
typedef int      i32x4 __attribute__((ext_vector_type(4)));
typedef int      i32x8 __attribute__((ext_vector_type(8)));

#define Bv    2048
#define Tv    512
#define TWINv 256
#define Hv    128

// workspace / LDS layout (bytes)
#define WS_FRAG   0         // 128 fragments * 512 halfs * 2B = 131072
#define WS_BIAS   131072    // 512 f32
#define WS_WIH    133120    // 512 f32
#define WS_WOUT   135168    // 128 f32
#define WS_TOTAL  135680    // bytes DMA'd into LDS (multiple of 16)
#define LDS_H     135680    // + wave*4096 : h buffer [16][128] f16
#define LDS_X     152064    // + wave*64   : x buffer [16] f32
#define LDS_TOTAL 152320

// ---------------------------------------------------------------------------
// Prep: pack W_hh into WMMA-B fragment layout (f16), fuse biases, copy W_ih/W_out
// Fragment f = (n_tile*4 + kc): 512 halfs, lane-major (32 lanes x 16 halfs).
// half j of lane l = W_hh[n][k], n = 16*n_tile + (l&15), k = 32*kc + 16*(l>>4) + j
// ---------------------------------------------------------------------------
__global__ void lstm_prep(const float* __restrict__ W_ih,
                          const float* __restrict__ W_hh,
                          const float* __restrict__ b_ih,
                          const float* __restrict__ b_hh,
                          const float* __restrict__ W_out,
                          unsigned char* __restrict__ ws) {
  _Float16* frag = (_Float16*)(ws + WS_FRAG);
  float* bias = (float*)(ws + WS_BIAS);
  float* wih  = (float*)(ws + WS_WIH);
  float* wout = (float*)(ws + WS_WOUT);
  int stride = gridDim.x * blockDim.x;
  for (int i = blockIdx.x * blockDim.x + threadIdx.x; i < 66688; i += stride) {
    if (i < 65536) {
      int f  = i >> 9;          // fragment index 0..127
      int w  = i & 511;
      int l  = w >> 4;          // lane 0..31
      int j  = w & 15;          // half 0..15
      int nt = f >> 2;          // n-tile 0..31
      int kc = f & 3;           // k-chunk 0..3
      int n  = nt * 16 + (l & 15);
      int k  = kc * 32 + 16 * (l >> 4) + j;
      frag[i] = (_Float16)W_hh[n * Hv + k];
    } else if (i < 66048) {
      int n = i - 65536; bias[n] = b_ih[n] + b_hh[n];
    } else if (i < 66560) {
      int n = i - 66048; wih[n] = W_ih[n];
    } else {
      int n = i - 66560; wout[n] = W_out[n];
    }
  }
}

// ---------------------------------------------------------------------------
// Recurrent kernel: 1 wave = 16 batch rows, 4 waves/block, 32 blocks.
// ---------------------------------------------------------------------------
__global__ void __launch_bounds__(128)
lstm_run(const float* __restrict__ y_flow,
         const unsigned char* __restrict__ ws,
         const float* __restrict__ b_out_p,
         float* __restrict__ out) {
  extern __shared__ unsigned char smem[];
  const int tid  = threadIdx.x;
  const int wave = tid >> 5;
  const int lane = tid & 31;
  const int g16  = lane >> 4;   // half-wave select (rows 0-7 vs 8-15 in D layout)
  const int lm   = lane & 15;

  // ---- TDM: pull packed weights (135,680 B) into LDS, wave 0 issues ----
  if (wave == 0) {
    unsigned long long gaddr = (unsigned long long)(uintptr_t)ws;
    unsigned int ldsa = (unsigned int)(uintptr_t)smem;   // LDS byte offset
    u32x4 g0;
    g0[0] = 1u;                                          // count=1, no gather
    g0[1] = ldsa;                                        // lds_addr
    g0[2] = (unsigned int)(gaddr & 0xFFFFFFFFull);       // global_addr lo
    g0[3] = (unsigned int)((gaddr >> 32) & 0x1FFFFFFu) | (2u << 30); // hi | type=2
    const unsigned int ndw = WS_TOTAL / 4;               // 33920 dwords
    i32x8 g1;
    g1[0] = (int)(2u << 16);                  // wg_mask=0, data_size=4B, no flags
    g1[1] = (int)((ndw & 0xFFFFu) << 16);     // tensor_dim0 lo16 (atomic_addr=0)
    g1[2] = (int)((ndw >> 16) | (1u << 16));  // tensor_dim0 hi | tensor_dim1=1 lo
    g1[3] = (int)((ndw & 0xFFFFu) << 16);     // tensor_dim1 hi=0 | tile_dim0
    g1[4] = 1;                                // tile_dim1=1, tile_dim2=0
    g1[5] = (int)ndw;                         // tensor_dim0_stride lo32
    g1[6] = 0;                                // stride0 hi | stride1 lo
    g1[7] = 0;
    i32x4 g2; g2[0]=0; g2[1]=0; g2[2]=0; g2[3]=0;
    i32x4 g3; g3[0]=0; g3[1]=0; g3[2]=0; g3[3]=0;
    i32x8 g4; g4[0]=0; g4[1]=0; g4[2]=0; g4[3]=0; g4[4]=0; g4[5]=0; g4[6]=0; g4[7]=0;
    __builtin_amdgcn_tensor_load_to_lds(g0, g1, g2, g3, g4, 0);
    __builtin_amdgcn_s_wait_tensorcnt(0);
  }

  // per-wave h scratch: zero it (h0 = 0) while TDM is in flight
  _Float16* hbuf = (_Float16*)(smem + LDS_H + wave * 4096);   // [16][128]
  float*    xbuf = (float*)(smem + LDS_X + wave * 64);        // [16]
  {
    unsigned int* hz = (unsigned int*)hbuf;
    #pragma unroll
    for (int i = 0; i < 32; ++i) hz[lane + 32 * i] = 0u;
  }
  __syncthreads();

  const _Float16* wfrag = (const _Float16*)(smem + WS_FRAG);
  const float* bias  = (const float*)(smem + WS_BIAS);
  const float* wih   = (const float*)(smem + WS_WIH);
  const float* woutl = (const float*)(smem + WS_WOUT);
  const float  bo    = b_out_p[0];
  const int    base  = blockIdx.x * 64 + wave * 16;

  float woutr[8];
  #pragma unroll
  for (int nt = 0; nt < 8; ++nt) woutr[nt] = woutl[nt * 16 + lm];

  // cell state in D layout: 8 column tiles x v8f
  v8f cst[8];
  #pragma unroll
  for (int i = 0; i < 8; ++i) {
    #pragma unroll
    for (int j = 0; j < 8; ++j) cst[i][j] = 0.0f;
  }

  for (int t = 0; t < Tv - 1; ++t) {
    // step input: teacher-forced flow, else previous prediction (already in xbuf)
    if (t < TWINv - 1) {
      if (lane < 16) xbuf[lane] = y_flow[(base + lane) * Tv + t];
    }
    float xv[8];
    #pragma unroll
    for (int r = 0; r < 8; ++r) xv[r] = xbuf[8 * g16 + r];

    // A fragments of h_{t-1}: 16x32 f16 per K-chunk, two contiguous b128 loads
    v16h a[4];
    #pragma unroll
    for (int kc = 0; kc < 4; ++kc) {
      const v8h* p = (const v8h*)(hbuf + lm * 128 + kc * 32 + 16 * g16);
      v8h lo = p[0];
      v8h hi = *(const v8h*)(hbuf + lm * 128 + kc * 32 + 16 * g16 + 8);
      #pragma unroll
      for (int j = 0; j < 8; ++j) { a[kc][j] = lo[j]; a[kc][j + 8] = hi[j]; }
    }

    float pacc[8];
    #pragma unroll
    for (int r = 0; r < 8; ++r) pacc[r] = 0.0f;

    #pragma unroll
    for (int nt = 0; nt < 8; ++nt) {          // hidden column tile
      v8f acc[4];
      #pragma unroll
      for (int gi = 0; gi < 4; ++gi) {        // gate i,f,g,o
        v8f z;
        #pragma unroll
        for (int j = 0; j < 8; ++j) z[j] = 0.0f;
        #pragma unroll
        for (int kc = 0; kc < 4; ++kc) {
          const int f = (gi * 8 + nt) * 4 + kc;
          const v8h* q = (const v8h*)(wfrag + f * 512 + lane * 16);
          v8h blo = q[0], bhi = q[1];
          v16h b;
          #pragma unroll
          for (int j = 0; j < 8; ++j) { b[j] = blo[j]; b[j + 8] = bhi[j]; }
          z = __builtin_amdgcn_wmma_f32_16x16x32_f16(
              false, a[kc], false, b, (short)0, z, false, false);
        }
        acc[gi] = z;
      }

      // elementwise gate math in D layout (lane -> column, vgpr -> row)
      const int ncol = nt * 16 + lm;
      const float bi  = bias[ncol],       wi = wih[ncol];
      const float bf  = bias[128 + ncol], wf = wih[128 + ncol];
      const float bg  = bias[256 + ncol], wg = wih[256 + ncol];
      const float bo_ = bias[384 + ncol], wo = wih[384 + ncol];
      const float won = woutr[nt];
      #pragma unroll
      for (int r = 0; r < 8; ++r) {
        float gi_v = acc[0][r] + bi  + xv[r] * wi;
        float gf_v = acc[1][r] + bf  + xv[r] * wf;
        float gg_v = acc[2][r] + bg  + xv[r] * wg;
        float go_v = acc[3][r] + bo_ + xv[r] * wo;
        float is = 1.0f / (1.0f + __expf(-gi_v));
        float fs = 1.0f / (1.0f + __expf(-gf_v));
        float e1 = __expf(-2.0f * gg_v);
        float gt = (1.0f - e1) / (1.0f + e1);
        float os = 1.0f / (1.0f + __expf(-go_v));
        float cn = fs * cst[nt][r] + is * gt;
        cst[nt][r] = cn;
        float e2 = __expf(-2.0f * cn);
        float ct = (1.0f - e2) / (1.0f + e2);
        float hv = os * ct;
        pacc[r] += hv * won;
        hbuf[(r + 8 * g16) * 128 + ncol] = (_Float16)hv;   // row-major for next A
      }
    }

    // prediction: butterfly-reduce over the 16 lanes of each half-wave
    #pragma unroll
    for (int r = 0; r < 8; ++r) {
      float v = pacc[r];
      v += __shfl_xor(v, 1, 32);
      v += __shfl_xor(v, 2, 32);
      v += __shfl_xor(v, 4, 32);
      v += __shfl_xor(v, 8, 32);
      pacc[r] = v + bo;
    }
    if (lm == 0) {
      #pragma unroll
      for (int r = 0; r < 8; ++r) {
        int m = r + 8 * g16;
        out[(size_t)(base + m) * (Tv - 1) + t] = pacc[r];
        xbuf[m] = pacc[r];   // autoregressive feedback
      }
    }
  }
}

extern "C" void kernel_launch(void* const* d_in, const int* in_sizes, int n_in,
                              void* d_out, int out_size, void* d_ws, size_t ws_size,
                              hipStream_t stream) {
  const float* y_flow = (const float*)d_in[0];
  const float* W_ih   = (const float*)d_in[2];
  const float* W_hh   = (const float*)d_in[3];
  const float* b_ih   = (const float*)d_in[4];
  const float* b_hh   = (const float*)d_in[5];
  const float* W_out  = (const float*)d_in[6];
  const float* b_out  = (const float*)d_in[7];
  float* out = (float*)d_out;
  unsigned char* ws = (unsigned char*)d_ws;

  lstm_prep<<<64, 256, 0, stream>>>(W_ih, W_hh, b_ih, b_hh, W_out, ws);
  lstm_run<<<32, 128, LDS_TOTAL, stream>>>(y_flow, ws, b_out, out);
}
